// SupervisedGNN_14955076125354
// MI455X (gfx1250) — compile-verified, matrix operand order checked by reading, not run on hardware
//
#include <hip/hip_runtime.h>
#include <hip/hip_bf16.h>

// ---------------- problem constants (match reference) ----------------
#define NN      50000
#define NE      800000
#define LAT     32
#define NSTEPS  10
#define NEG_SLOPE 0.01f
#define LN_EPS  1e-5f

#define EPB 128   // edges per block (8 wave32s x 16-row tiles)
#define NPB 128   // nodes per block

// Padded LDS pitches (bank-conflict-free on 64 x 4B banks):
//  f16 tiles: 40 halves/row (80 B, 16B-aligned, row starts distinct mod 64 banks)
//  f32 tiles: 36 floats/row (144 B, 16B-aligned, stride 36 mod 64 distinct)
#define HP 40
#define FP 36

typedef __attribute__((ext_vector_type(16))) _Float16 v16h;
typedef __attribute__((ext_vector_type(8)))  _Float16 v8h;
typedef __attribute__((ext_vector_type(8)))  float    v8f;

__device__ __forceinline__ float leakyf(float v) { return v >= 0.f ? v : NEG_SLOPE * v; }

// Build a 16x32 f16 A-fragment row slice for this lane from an LDS row.
// Per CDNA5 WMMA A layout: lane<16 holds K {0..7,16..23}, lane>=16 holds K {8..15,24..31}.
__device__ __forceinline__ v16h load_a_frag(const _Float16* rowp, int kb) {
  v8h lo = *(const v8h*)(rowp + kb);        // K = kb .. kb+7
  v8h hi = *(const v8h*)(rowp + 16 + kb);   // K = 16+kb .. 16+kb+7
  return __builtin_shufflevector(lo, hi, 0, 1, 2, 3, 4, 5, 6, 7,
                                         8, 9, 10, 11, 12, 13, 14, 15);
}

__device__ __forceinline__ v8h pack8(float4 a, float4 b) {
  v8h r = { (_Float16)a.x, (_Float16)a.y, (_Float16)a.z, (_Float16)a.w,
            (_Float16)b.x, (_Float16)b.y, (_Float16)b.z, (_Float16)b.w };
  return r;
}

// ---------------------------------------------------------------------
// Pack fp32 weights into f16 WMMA B-fragments, one v16h per lane.
// Fragment index layout: [step][kchunk][nhalf][lane][16 halves]
// B (K x N) mirror of A layout: lane<16 -> col N=lane, K {0..7,16..23};
// lane>=16 -> col N=lane-16, K {8..15,24..31}  (within a 32-K chunk).
// ---------------------------------------------------------------------
__global__ __launch_bounds__(256) void pack_weights_kernel(
    const float* __restrict__ eW,   // [NSTEPS][96][32]
    const float* __restrict__ nW,   // [NSTEPS][64][32]
    _Float16* __restrict__ pE,      // [NSTEPS*3*2*32][16]
    _Float16* __restrict__ pN)      // [NSTEPS*2*2*32][16]
{
  const int tid = blockIdx.x * blockDim.x + threadIdx.x;
  const int ETASKS = NSTEPS * 3 * 2 * 32;
  const int NTASKS = NSTEPS * 2 * 2 * 32;
  if (tid < ETASKS) {
    const int l  = tid & 31;
    const int t2 = tid >> 5;
    const int hN = t2 & 1;
    const int t3 = t2 >> 1;
    const int c  = t3 % 3;
    const int s  = t3 / 3;
    const int n  = hN * 16 + (l & 15);
    const int kb = (l >> 4) * 8;
#pragma unroll
    for (int j = 0; j < 16; ++j) {
      const int lk = (j < 8) ? (kb + j) : (16 + kb + (j - 8));
      const int k  = c * 32 + lk;
      pE[(size_t)tid * 16 + j] = (_Float16)eW[((size_t)s * 96 + k) * 32 + n];
    }
  } else if (tid < ETASKS + NTASKS) {
    const int T  = tid - ETASKS;
    const int l  = T & 31;
    const int t2 = T >> 5;
    const int hN = t2 & 1;
    const int t3 = t2 >> 1;
    const int c  = t3 & 1;
    const int s  = t3 >> 1;
    const int n  = hN * 16 + (l & 15);
    const int kb = (l >> 4) * 8;
#pragma unroll
    for (int j = 0; j < 16; ++j) {
      const int lk = (j < 8) ? (kb + j) : (16 + kb + (j - 8));
      const int k  = c * 32 + lk;
      pN[(size_t)T * 16 + j] = (_Float16)nW[((size_t)s * 64 + k) * 32 + n];
    }
  }
}

// ---------------- tiny encoders / utility kernels --------------------
__global__ __launch_bounds__(256) void encode_nodes_kernel(
    const float* __restrict__ x, const float* __restrict__ W,
    const float* __restrict__ b, float* __restrict__ h)
{
  __shared__ float sW[6 * LAT];
  __shared__ float sb[LAT];
  const int tid = threadIdx.x;
  if (tid < 6 * LAT) sW[tid] = W[tid];
  if (tid < LAT)     sb[tid] = b[tid];
  __syncthreads();
  const int n = blockIdx.x * blockDim.x + tid;
  if (n >= NN) return;
  float xi[6];
#pragma unroll
  for (int i = 0; i < 6; ++i) xi[i] = x[(size_t)n * 6 + i];
#pragma unroll
  for (int o = 0; o < LAT; ++o) {
    float acc = sb[o];
#pragma unroll
    for (int i = 0; i < 6; ++i) acc += xi[i] * sW[i * LAT + o];
    h[(size_t)n * LAT + o] = leakyf(acc);
  }
}

__global__ __launch_bounds__(256) void encode_edges_kernel(
    const float* __restrict__ ea, const float* __restrict__ W,
    const float* __restrict__ b, float* __restrict__ e)
{
  __shared__ float sW[3 * LAT];
  __shared__ float sb[LAT];
  const int tid = threadIdx.x;
  if (tid < 3 * LAT) sW[tid] = W[tid];
  if (tid < LAT)     sb[tid] = b[tid];
  __syncthreads();
  const int n = blockIdx.x * blockDim.x + tid;
  if (n >= NE) return;
  float xi[3];
#pragma unroll
  for (int i = 0; i < 3; ++i) xi[i] = ea[(size_t)n * 3 + i];
#pragma unroll
  for (int o = 0; o < LAT; ++o) {
    float acc = sb[o];
#pragma unroll
    for (int i = 0; i < 3; ++i) acc += xi[i] * sW[i * LAT + o];
    e[(size_t)n * LAT + o] = leakyf(acc);
  }
}

__global__ __launch_bounds__(256) void zero_f32_kernel(float* __restrict__ p, int n) {
  const int i = blockIdx.x * blockDim.x + threadIdx.x;
  if (i < n) p[i] = 0.f;
}

__global__ __launch_bounds__(256) void count_kernel(
    const int* __restrict__ recv, float* __restrict__ cnt)
{
  const int i = blockIdx.x * blockDim.x + threadIdx.x;
  if (i < NE) unsafeAtomicAdd(&cnt[recv[i]], 1.0f);
}

__global__ __launch_bounds__(256) void invcnt_kernel(
    const float* __restrict__ cnt, float* __restrict__ inv)
{
  const int i = blockIdx.x * blockDim.x + threadIdx.x;
  if (i < NN) inv[i] = 1.0f / fmaxf(cnt[i], 1.0f);
}

// ---------------------------------------------------------------------
// Edge update: m = LN(leaky([e | h_src | h_dst] @ W + b)); e += m;
// then scatter-add new e into agg[receiver].
// 128 edges/block; each wave owns a 16x96 A tile -> 6 WMMAs.
// Old e is re-read from global in the epilogue (e is L2-resident: 102 MB
// working set vs 192 MB L2), which frees LDS for conflict-free padding.
// ---------------------------------------------------------------------
__global__ __launch_bounds__(256) void edge_update_kernel(
    float* __restrict__ e, const float* __restrict__ h,
    const int* __restrict__ senders, const int* __restrict__ receivers,
    const v16h* __restrict__ Bfrag,   // 6 fragments x 32 lanes (this step)
    const float* __restrict__ eb, const float* __restrict__ es,
    const float* __restrict__ ebi, float* __restrict__ agg)
{
  __shared__ _Float16 sE [EPB][HP];
  __shared__ _Float16 sHs[EPB][HP];
  __shared__ _Float16 sHd[EPB][HP];
  __shared__ float    sD [EPB][FP];
  __shared__ int      sRecv[EPB];
  __shared__ float    sEB[LAT], sES[LAT], sEBI[LAT];

  const int  tid  = threadIdx.x;
  const long base = (long)blockIdx.x * EPB;

  if (tid < LAT) { sEB[tid] = eb[tid]; sES[tid] = es[tid]; sEBI[tid] = ebi[tid]; }

  // ---- stage: e tile and gathered h_src/h_dst tiles (f16, b128 stores)
  {
    const int  eloc = tid >> 1;
    const int  c0   = (tid & 1) * 16;
    const long eg   = base + eloc;           // NE % EPB == 0: always valid
    const int  sidx = senders[eg];
    const int  ridx = receivers[eg];
    if (c0 == 0) sRecv[eloc] = ridx;
    const float4* ep = (const float4*)(e + eg * LAT + c0);
    const float4* sp = (const float4*)(h + (long)sidx * LAT + c0);
    const float4* rp = (const float4*)(h + (long)ridx * LAT + c0);
    const float4 e0 = ep[0], e1 = ep[1], e2 = ep[2], e3 = ep[3];
    const float4 s0 = sp[0], s1 = sp[1], s2 = sp[2], s3 = sp[3];
    const float4 r0 = rp[0], r1 = rp[1], r2 = rp[2], r3 = rp[3];
    *(v8h*)&sE [eloc][c0]     = pack8(e0, e1);
    *(v8h*)&sE [eloc][c0 + 8] = pack8(e2, e3);
    *(v8h*)&sHs[eloc][c0]     = pack8(s0, s1);
    *(v8h*)&sHs[eloc][c0 + 8] = pack8(s2, s3);
    *(v8h*)&sHd[eloc][c0]     = pack8(r0, r1);
    *(v8h*)&sHd[eloc][c0 + 8] = pack8(r2, r3);
    // speculative prefetch of the next block's e tile (kept L2/WGP hot)
    if (base + EPB < NE)
      __builtin_prefetch(e + (base + EPB) * LAT + (long)tid * 16, 0, 3);
  }
  __syncthreads();

  // ---- WMMA: D(16x32) = A(16x96) x B(96x32), fp32 accum
  const int wave = tid >> 5;
  const int lane = tid & 31;
  const int row  = (wave << 4) + (lane & 15);
  const int kb   = (lane >> 4) * 8;

  const v16h b00 = Bfrag[0 * 32 + lane];
  const v16h b01 = Bfrag[1 * 32 + lane];
  const v16h b10 = Bfrag[2 * 32 + lane];
  const v16h b11 = Bfrag[3 * 32 + lane];
  const v16h b20 = Bfrag[4 * 32 + lane];
  const v16h b21 = Bfrag[5 * 32 + lane];

  v8f acc0 = {};
  v8f acc1 = {};
  {
    v16h a = load_a_frag(&sE[row][0], kb);
    acc0 = __builtin_amdgcn_wmma_f32_16x16x32_f16(false, a, false, b00, (short)0, acc0, false, false);
    acc1 = __builtin_amdgcn_wmma_f32_16x16x32_f16(false, a, false, b01, (short)0, acc1, false, false);
  }
  {
    v16h a = load_a_frag(&sHs[row][0], kb);
    acc0 = __builtin_amdgcn_wmma_f32_16x16x32_f16(false, a, false, b10, (short)0, acc0, false, false);
    acc1 = __builtin_amdgcn_wmma_f32_16x16x32_f16(false, a, false, b11, (short)0, acc1, false, false);
  }
  {
    v16h a = load_a_frag(&sHd[row][0], kb);
    acc0 = __builtin_amdgcn_wmma_f32_16x16x32_f16(false, a, false, b20, (short)0, acc0, false, false);
    acc1 = __builtin_amdgcn_wmma_f32_16x16x32_f16(false, a, false, b21, (short)0, acc1, false, false);
  }

  // ---- bias + leaky, spill D tile to LDS (C/D layout: vgpr g=row, lane=col)
  {
    const int col = lane & 15;
    const int r0  = (wave << 4) + ((lane >> 4) << 3);
    const float bc0 = sEB[col], bc1 = sEB[16 + col];
#pragma unroll
    for (int g = 0; g < 8; ++g) {
      sD[r0 + g][col]      = leakyf(acc0[g] + bc0);
      sD[r0 + g][16 + col] = leakyf(acc1[g] + bc1);
    }
  }
  __syncthreads();

  // ---- LayerNorm + residual + scatter-add; 2 lanes per row (16 feats each)
  {
    const int rloc = (wave << 4) + (lane & 15);
    const int fb   = (lane >> 4) << 4;   // 0 or 16
    float d[16];
    float sum = 0.f, sq = 0.f;
#pragma unroll
    for (int j = 0; j < 16; ++j) { const float v = sD[rloc][fb + j]; d[j] = v; sum += v; sq += v * v; }
    sum += __shfl_xor(sum, 16, 32);
    sq  += __shfl_xor(sq, 16, 32);
    const float mu   = sum * (1.f / 32.f);
    const float var  = sq * (1.f / 32.f) - mu * mu;
    const float rstd = rsqrtf(var + LN_EPS);
    const long eg   = base + rloc;
    const int  ridx = sRecv[rloc];
    float* erow = e + eg * LAT + fb;
    float* arow = agg + (long)ridx * LAT + fb;
    // re-read old e row-half from global (L2 hit) for the residual
    const float4* eoldp = (const float4*)erow;
    const float4 o0 = eoldp[0], o1 = eoldp[1], o2 = eoldp[2], o3 = eoldp[3];
    const float eo[16] = { o0.x, o0.y, o0.z, o0.w, o1.x, o1.y, o1.z, o1.w,
                           o2.x, o2.y, o2.z, o2.w, o3.x, o3.y, o3.z, o3.w };
    float ov[16];
#pragma unroll
    for (int j = 0; j < 16; ++j) {
      const int f = fb + j;
      ov[j] = eo[j] + (d[j] - mu) * rstd * sES[f] + sEBI[f];
    }
    float4* ew = (float4*)erow;
    ew[0] = make_float4(ov[0],  ov[1],  ov[2],  ov[3]);
    ew[1] = make_float4(ov[4],  ov[5],  ov[6],  ov[7]);
    ew[2] = make_float4(ov[8],  ov[9],  ov[10], ov[11]);
    ew[3] = make_float4(ov[12], ov[13], ov[14], ov[15]);
#pragma unroll
    for (int j = 0; j < 16; ++j) unsafeAtomicAdd(&arow[j], ov[j]);
  }
}

// ---------------------------------------------------------------------
// Node update: u = LN(leaky([h | agg*inv] @ W + b)); h += u.
// 128 nodes/block; each wave owns a 16x64 A tile -> 4 WMMAs.
// ---------------------------------------------------------------------
__global__ __launch_bounds__(256) void node_update_kernel(
    float* __restrict__ h, const float* __restrict__ agg,
    const float* __restrict__ inv,
    const v16h* __restrict__ Bfrag,   // 4 fragments x 32 lanes (this step)
    const float* __restrict__ nb, const float* __restrict__ ns,
    const float* __restrict__ nbi)
{
  __shared__ _Float16 sH[NPB][HP];
  __shared__ _Float16 sA[NPB][HP];
  __shared__ float    sD[NPB][FP];
  __shared__ float    sNB[LAT], sNS[LAT], sNBI[LAT];

  const int  tid  = threadIdx.x;
  const long base = (long)blockIdx.x * NPB;

  if (tid < LAT) { sNB[tid] = nb[tid]; sNS[tid] = ns[tid]; sNBI[tid] = nbi[tid]; }

  {
    const int  loc = tid >> 1;
    const int  c0  = (tid & 1) * 16;
    const long ng  = base + loc;
    const long nc  = (ng < NN) ? ng : (NN - 1);   // clamp tail loads
    const float ic = inv[nc];
    const float4* hp = (const float4*)(h   + nc * LAT + c0);
    const float4* ap = (const float4*)(agg + nc * LAT + c0);
    const float4 h0 = hp[0], h1 = hp[1], h2 = hp[2], h3 = hp[3];
    float4 a0 = ap[0], a1 = ap[1], a2 = ap[2], a3 = ap[3];
    a0.x *= ic; a0.y *= ic; a0.z *= ic; a0.w *= ic;
    a1.x *= ic; a1.y *= ic; a1.z *= ic; a1.w *= ic;
    a2.x *= ic; a2.y *= ic; a2.z *= ic; a2.w *= ic;
    a3.x *= ic; a3.y *= ic; a3.z *= ic; a3.w *= ic;
    *(v8h*)&sH[loc][c0]     = pack8(h0, h1);
    *(v8h*)&sH[loc][c0 + 8] = pack8(h2, h3);
    *(v8h*)&sA[loc][c0]     = pack8(a0, a1);
    *(v8h*)&sA[loc][c0 + 8] = pack8(a2, a3);
  }
  __syncthreads();

  const int wave = tid >> 5;
  const int lane = tid & 31;
  const int row  = (wave << 4) + (lane & 15);
  const int kb   = (lane >> 4) * 8;

  const v16h b00 = Bfrag[0 * 32 + lane];
  const v16h b01 = Bfrag[1 * 32 + lane];
  const v16h b10 = Bfrag[2 * 32 + lane];
  const v16h b11 = Bfrag[3 * 32 + lane];

  v8f acc0 = {};
  v8f acc1 = {};
  {
    v16h a = load_a_frag(&sH[row][0], kb);
    acc0 = __builtin_amdgcn_wmma_f32_16x16x32_f16(false, a, false, b00, (short)0, acc0, false, false);
    acc1 = __builtin_amdgcn_wmma_f32_16x16x32_f16(false, a, false, b01, (short)0, acc1, false, false);
  }
  {
    v16h a = load_a_frag(&sA[row][0], kb);
    acc0 = __builtin_amdgcn_wmma_f32_16x16x32_f16(false, a, false, b10, (short)0, acc0, false, false);
    acc1 = __builtin_amdgcn_wmma_f32_16x16x32_f16(false, a, false, b11, (short)0, acc1, false, false);
  }

  {
    const int col = lane & 15;
    const int r0  = (wave << 4) + ((lane >> 4) << 3);
    const float bc0 = sNB[col], bc1 = sNB[16 + col];
#pragma unroll
    for (int g = 0; g < 8; ++g) {
      sD[r0 + g][col]      = leakyf(acc0[g] + bc0);
      sD[r0 + g][16 + col] = leakyf(acc1[g] + bc1);
    }
  }
  __syncthreads();

  {
    const int rloc = (wave << 4) + (lane & 15);
    const int fb   = (lane >> 4) << 4;
    float d[16];
    float sum = 0.f, sq = 0.f;
#pragma unroll
    for (int j = 0; j < 16; ++j) { const float v = sD[rloc][fb + j]; d[j] = v; sum += v; sq += v * v; }
    sum += __shfl_xor(sum, 16, 32);
    sq  += __shfl_xor(sq, 16, 32);
    const float mu   = sum * (1.f / 32.f);
    const float var  = sq * (1.f / 32.f) - mu * mu;
    const float rstd = rsqrtf(var + LN_EPS);
    const long ng = base + rloc;
    if (ng < NN) {
      float* hrow = h + ng * LAT + fb;
      const float4* holdp = (const float4*)hrow;
      const float4 o0 = holdp[0], o1 = holdp[1], o2 = holdp[2], o3 = holdp[3];
      const float ho[16] = { o0.x, o0.y, o0.z, o0.w, o1.x, o1.y, o1.z, o1.w,
                             o2.x, o2.y, o2.z, o2.w, o3.x, o3.y, o3.z, o3.w };
      float ov[16];
#pragma unroll
      for (int j = 0; j < 16; ++j) {
        const int f = fb + j;
        ov[j] = ho[j] + (d[j] - mu) * rstd * sNS[f] + sNBI[f];
      }
      float4* hw = (float4*)hrow;
      hw[0] = make_float4(ov[0],  ov[1],  ov[2],  ov[3]);
      hw[1] = make_float4(ov[4],  ov[5],  ov[6],  ov[7]);
      hw[2] = make_float4(ov[8],  ov[9],  ov[10], ov[11]);
      hw[3] = make_float4(ov[12], ov[13], ov[14], ov[15]);
    }
  }
}

// ---------------- decoder: leaky(h@W1+b1)@W2+b2 ----------------------
__global__ __launch_bounds__(256) void decoder_kernel(
    const float* __restrict__ h,
    const float* __restrict__ W1, const float* __restrict__ b1,
    const float* __restrict__ W2, const float* __restrict__ b2,
    float* __restrict__ out)
{
  __shared__ float sW1[LAT * LAT];
  __shared__ float sW2[LAT];
  __shared__ float sb1[LAT];
  const int tid = threadIdx.x;
#pragma unroll
  for (int q = 0; q < 4; ++q) sW1[tid * 4 + q] = W1[tid * 4 + q];
  if (tid < LAT) { sW2[tid] = W2[tid]; sb1[tid] = b1[tid]; }
  __syncthreads();
  const int n = blockIdx.x * blockDim.x + tid;
  if (n >= NN) return;
  float hr[LAT];
  const float4* hp = (const float4*)(h + (size_t)n * LAT);
#pragma unroll
  for (int q = 0; q < 8; ++q) {
    const float4 v = hp[q];
    hr[q * 4 + 0] = v.x; hr[q * 4 + 1] = v.y; hr[q * 4 + 2] = v.z; hr[q * 4 + 3] = v.w;
  }
  float oacc = 0.f;
#pragma unroll
  for (int o = 0; o < LAT; ++o) {
    float acc = sb1[o];
#pragma unroll
    for (int i = 0; i < LAT; ++i) acc += hr[i] * sW1[i * LAT + o];
    oacc += leakyf(acc) * sW2[o];
  }
  out[n] = oacc + b2[0];
}

// ---------------------------------------------------------------------
extern "C" void kernel_launch(void* const* d_in, const int* in_sizes, int n_in,
                              void* d_out, int out_size, void* d_ws, size_t ws_size,
                              hipStream_t stream) {
  (void)in_sizes; (void)n_in; (void)out_size; (void)ws_size;

  const float* x          = (const float*)d_in[0];
  const float* edge_attr  = (const float*)d_in[1];
  const int*   senders    = (const int*)d_in[2];
  const int*   receivers  = (const int*)d_in[3];
  const float* node_enc_W = (const float*)d_in[4];
  const float* node_enc_b = (const float*)d_in[5];
  const float* edge_enc_W = (const float*)d_in[6];
  const float* edge_enc_b = (const float*)d_in[7];
  const float* edge_W     = (const float*)d_in[8];
  const float* edge_b     = (const float*)d_in[9];
  const float* edge_ls    = (const float*)d_in[10];
  const float* edge_lb    = (const float*)d_in[11];
  const float* node_W     = (const float*)d_in[12];
  const float* node_b     = (const float*)d_in[13];
  const float* node_ls    = (const float*)d_in[14];
  const float* node_lb    = (const float*)d_in[15];
  const float* dec_W1     = (const float*)d_in[16];
  const float* dec_b1     = (const float*)d_in[17];
  const float* dec_W2     = (const float*)d_in[18];
  const float* dec_b2     = (const float*)d_in[19];
  float* out = (float*)d_out;

  // workspace carve-up (256B aligned)
  char* ws = (char*)d_ws;
  size_t off = 0;
  auto take = [&](size_t bytes) -> char* {
    char* p = ws + off;
    off += (bytes + 255) & ~(size_t)255;
    return p;
  };
  float*    h    = (float*)take((size_t)NN * LAT * sizeof(float));
  float*    e    = (float*)take((size_t)NE * LAT * sizeof(float));
  float*    agg  = (float*)take((size_t)NN * LAT * sizeof(float));
  float*    cnt  = (float*)take((size_t)NN * sizeof(float));
  float*    inv  = (float*)take((size_t)NN * sizeof(float));
  _Float16* pE   = (_Float16*)take((size_t)NSTEPS * 3 * 2 * 32 * 16 * sizeof(_Float16));
  _Float16* pN   = (_Float16*)take((size_t)NSTEPS * 2 * 2 * 32 * 16 * sizeof(_Float16));
  const v16h* pEv = (const v16h*)pE;
  const v16h* pNv = (const v16h*)pN;

  // 1) pack weights into WMMA B fragments (f32 -> f16, lane-swizzled)
  {
    const int tasks = NSTEPS * 3 * 2 * 32 + NSTEPS * 2 * 2 * 32;
    pack_weights_kernel<<<(tasks + 255) / 256, 256, 0, stream>>>(edge_W, node_W, pE, pN);
  }
  // 2) encoders
  encode_nodes_kernel<<<(NN + 255) / 256, 256, 0, stream>>>(x, node_enc_W, node_enc_b, h);
  encode_edges_kernel<<<(NE + 255) / 256, 256, 0, stream>>>(edge_attr, edge_enc_W, edge_enc_b, e);
  // 3) in-degree -> inverse counts
  zero_f32_kernel<<<(NN + 255) / 256, 256, 0, stream>>>(cnt, NN);
  count_kernel<<<(NE + 255) / 256, 256, 0, stream>>>(receivers, cnt);
  invcnt_kernel<<<(NN + 255) / 256, 256, 0, stream>>>(cnt, inv);

  // 4) message-passing steps
  for (int s = 0; s < NSTEPS; ++s) {
    zero_f32_kernel<<<(NN * LAT + 255) / 256, 256, 0, stream>>>(agg, NN * LAT);
    edge_update_kernel<<<NE / EPB, 256, 0, stream>>>(
        e, h, senders, receivers,
        pEv + (size_t)s * 6 * 32,
        edge_b + (size_t)s * LAT, edge_ls + (size_t)s * LAT, edge_lb + (size_t)s * LAT,
        agg);
    node_update_kernel<<<(NN + NPB - 1) / NPB, 256, 0, stream>>>(
        h, agg, inv,
        pNv + (size_t)s * 4 * 32,
        node_b + (size_t)s * LAT, node_ls + (size_t)s * LAT, node_lb + (size_t)s * LAT);
  }

  // 5) decoder
  decoder_kernel<<<(NN + 255) / 256, 256, 0, stream>>>(h, dec_W1, dec_b1, dec_W2, dec_b2, out);
}
